// MyQuantizedLinear_60498909331538
// MI455X (gfx1250) — 8-bit matrix operand order repaired and checked by reading, compiled, not run on hardware
//
#include <hip/hip_runtime.h>
#include <stdint.h>

typedef int v8i __attribute__((ext_vector_type(8)));

#define BATCH 32768
#define KDIM  1024
#define NDIM  1024

#define MT 128
#define NT 128
#define KT 64
#define LDS_STRIDE 80
#define BUF_BYTES ((MT + NT) * LDS_STRIDE)
#define NBUF 3

__device__ __forceinline__ void async_copy16s(uint32_t lds_addr, uint32_t voff,
                                              const uint8_t* sbase) {
    asm volatile("global_load_async_to_lds_b128 %0, %1, %2"
                 :: "v"(lds_addr), "v"(voff), "s"(sbase) : "memory");
}
template <int N>
__device__ __forceinline__ void wait_async() {
    asm volatile("s_wait_asynccnt %0" :: "i"(N) : "memory");
}

__global__ __launch_bounds__(256) void pack_rowsum(const int* __restrict__ src,
                                                   uint32_t* __restrict__ dst,
                                                   int* __restrict__ sums) {
    __shared__ int red[256];
    const int row = blockIdx.x;
    const int t = threadIdx.x;
    const int4 s = ((const int4*)src)[(size_t)row * 256 + t];
    uint32_t p = (uint32_t)(s.x & 0xFF) |
                 ((uint32_t)(s.y & 0xFF) << 8) |
                 ((uint32_t)(s.z & 0xFF) << 16) |
                 ((uint32_t)(s.w & 0xFF) << 24);
    dst[(size_t)row * 256 + t] = p;
    red[t] = s.x + s.y + s.z + s.w;
    __syncthreads();
    for (int off = 128; off > 0; off >>= 1) {
        if (t < off) red[t] += red[t + off];
        __syncthreads();
    }
    if (t == 0) sums[row] = red[0];
}

__device__ __forceinline__ int rdhm(int a, int mult) {
    int64_t ab = (int64_t)a * (int64_t)mult;
    int64_t nudge = (ab >= 0) ? (int64_t)(1ll << 30) : (int64_t)(1ll - (1ll << 30));
    int64_t res = (ab + nudge) >> 31;
    res = res > 2147483647ll ? 2147483647ll : res;
    res = res < -2147483648ll ? -2147483648ll : res;
    return (int)res;
}

__device__ __forceinline__ int rrshift(int x, int sh) {
    if (sh == 0) return x;
    int mask = (1 << sh) - 1;
    int rem = x & mask;
    int thr = (mask >> 1) + ((x < 0) ? 1 : 0);
    return (x >> sh) + ((rem > thr) ? 1 : 0);
}

__global__ __launch_bounds__(256) void qgemm_wmma(
    const uint8_t* __restrict__ Xp, const uint8_t* __restrict__ Wp,
    const int* __restrict__ xsum, const int* __restrict__ wsum,
    const int* __restrict__ qbias,
    const int* __restrict__ p_izp, const int* __restrict__ p_wzp,
    const int* __restrict__ p_mult, const int* __restrict__ p_shift,
    const int* __restrict__ p_ozp,
    float* __restrict__ out) {

    __shared__ __align__(16) uint8_t lds[NBUF * BUF_BYTES];

    const int t    = threadIdx.x;
    const int lane = t & 31;
    const int wave = t >> 5;
    const int wm   = wave & 3;
    const int wn   = wave >> 2;
    const int mbase = blockIdx.y * MT;
    const int nbase = blockIdx.x * NT;

    const int hl = lane >> 4;
    const int ml = lane & 15;

    const v8i vzero = {0, 0, 0, 0, 0, 0, 0, 0};
    v8i acc[2][4];
#pragma unroll
    for (int i = 0; i < 2; i++)
#pragma unroll
        for (int j = 0; j < 4; j++) acc[i][j] = vzero;

    const int grow = t >> 2;
    const int gcol = (t & 3) * 16;
    const uint32_t lds_base = (uint32_t)(uintptr_t)&lds[0];

    const uint32_t voff0 = (uint32_t)(grow * KDIM + gcol);
    const uint32_t voff1 = (uint32_t)((64 + grow) * KDIM + gcol);
    const uint8_t* xb = Xp + (size_t)mbase * KDIM;
    const uint8_t* wb = Wp + (size_t)nbase * KDIM;

    uint32_t lA[NBUF][2], lB[NBUF][2];
#pragma unroll
    for (int b = 0; b < NBUF; b++)
#pragma unroll
        for (int j = 0; j < 2; j++) {
            lA[b][j] = lds_base + b * BUF_BYTES + (j * 64 + grow) * LDS_STRIDE + gcol;
            lB[b][j] = lA[b][j] + MT * LDS_STRIDE;
        }

    auto stage = [&](int slab) {
        const int b = slab % NBUF;
        const uint8_t* xk = xb + slab * KT;
        const uint8_t* wk = wb + slab * KT;
        async_copy16s(lA[b][0], voff0, xk);
        async_copy16s(lB[b][0], voff0, wk);
        async_copy16s(lA[b][1], voff1, xk);
        async_copy16s(lB[b][1], voff1, wk);
    };

    const int NITER = KDIM / KT;
    stage(0);
    stage(1);

#pragma unroll
    for (int it = 0; it < NITER; ++it) {
        if (it + 1 < NITER) wait_async<4>(); else wait_async<0>();
        __syncthreads();
        if (it + 2 < NITER) {
            stage(it + 2);
            __builtin_prefetch(Xp + (size_t)(mbase + grow) * KDIM + (it + 4) * KT + gcol, 0, 1);
            __builtin_prefetch(Wp + (size_t)(nbase + grow) * KDIM + (it + 4) * KT + gcol, 0, 1);
        }
        const uint8_t* curA = &lds[(it % NBUF) * BUF_BYTES];
        const uint8_t* curB = curA + MT * LDS_STRIDE;

        union { v8i v; uint64_t q[4]; } af[2];
#pragma unroll
        for (int mi = 0; mi < 2; mi++) {
            const int arow = wm * 32 + mi * 16 + ml;
#pragma unroll
            for (int c = 0; c < 4; c++) {
                af[mi].q[c] =
                    *(const uint64_t*)(curA + arow * LDS_STRIDE + (c >> 1) * 32 + hl * 16 + (c & 1) * 8);
            }
        }
        union { v8i v; int4 x[2]; } bf[4];
#pragma unroll
        for (int ni = 0; ni < 4; ni++) {
            const int brow = wn * 64 + ni * 16 + ml;
#pragma unroll
            for (int g = 0; g < 2; g++) {
                bf[ni].x[g] =
                    *(const int4*)(curB + brow * LDS_STRIDE + g * 32 + hl * 16);
            }
        }

#pragma unroll
        for (int mi = 0; mi < 2; mi++)
#pragma unroll
            for (int ni = 0; ni < 4; ni++)
                acc[mi][ni] = __builtin_amdgcn_wmma_i32_16x16x64_iu8(
                    true, af[mi].v, true, bf[ni].v, acc[mi][ni], false, false);
    }

    const int izp = *p_izp, wzp = *p_wzp, mult = *p_mult;
    const int shift = *p_shift, ozp = *p_ozp;
    const int constoff = izp * wzp * KDIM;

#pragma unroll
    for (int ni = 0; ni < 4; ni++) {
        const int gn = nbase + wn * 64 + ni * 16 + ml;
        const int corr = qbias[gn] - wsum[gn] * izp + constoff;
#pragma unroll
        for (int mi = 0; mi < 2; mi++) {
            const int rowb = mbase + wm * 32 + mi * 16 + hl * 8;
#pragma unroll
            for (int r = 0; r < 8; r++) {
                const int gm = rowb + r;
                int z = acc[mi][ni][r] + corr - xsum[gm] * wzp;
                z = rrshift(rdhm(z, mult), shift);
                z += ozp;
                z = z > 127 ? 127 : (z < -127 ? -127 : z);
                out[(size_t)gm * NDIM + gn] = (float)z;
            }
        }
    }
}

extern "C" void kernel_launch(void* const* d_in, const int* in_sizes, int n_in,
                              void* d_out, int out_size, void* d_ws, size_t ws_size,
                              hipStream_t stream) {
    const int* x_i32   = (const int*)d_in[0];
    const int* w_i32   = (const int*)d_in[1];
    const int* qbias   = (const int*)d_in[2];
    const int* p_izp   = (const int*)d_in[3];
    const int* p_wzp   = (const int*)d_in[4];
    const int* p_mult  = (const int*)d_in[5];
    const int* p_shift = (const int*)d_in[6];
    const int* p_ozp   = (const int*)d_in[7];
    float* out = (float*)d_out;

    uint8_t* ws = (uint8_t*)d_ws;
    uint8_t* Xp = ws;
    uint8_t* Wp = ws + (size_t)BATCH * KDIM;
    int* xsum = (int*)(ws + (size_t)BATCH * KDIM + (size_t)NDIM * KDIM);
    int* wsum = (int*)((uint8_t*)xsum + (size_t)BATCH * sizeof(int));

    pack_rowsum<<<BATCH, 256, 0, stream>>>(x_i32, (uint32_t*)Xp, xsum);
    pack_rowsum<<<NDIM, 256, 0, stream>>>(w_i32, (uint32_t*)Wp, wsum);

    dim3 grid(NDIM / NT, BATCH / MT);
    qgemm_wmma<<<grid, 256, 0, stream>>>(Xp, Wp, xsum, wsum, qbias,
                                         p_izp, p_wzp, p_mult, p_shift, p_ozp, out);
}
